// MoEPolicyNet_3229815406996
// MI455X (gfx1250) — matrix operand order repaired and checked by reading, not compiled, not run on hardware
//
#include <hip/hip_runtime.h>
#include <hip/hip_bf16.h>

typedef __attribute__((ext_vector_type(16))) __bf16 v16bf;
typedef __attribute__((ext_vector_type(8)))  __bf16 v8bf;
typedef __attribute__((ext_vector_type(8)))  float  v8f;

#define BATCH   131072
#define FDIM    512
#define NEXP    4
#define HDIM    256
#define ADIM    32
#define M_TILE  128
#define XS_STRIDE 520   // 512 + 8 bf16 pad (row stride 1040B, 16B aligned)
#define HS_STRIDE 264   // 256 + 8 bf16 pad (528B, 16B aligned)

// ---------------- prep: transpose + convert weights to bf16 in workspace ----
// W1t[e][h][f] = bf16(W1[e][f][h]);  W2t[e][a][h] = bf16(W2[e][h][a])
__global__ void moe_prep_weights(const float* __restrict__ W1,
                                 const float* __restrict__ W2,
                                 __bf16* __restrict__ W1t,
                                 __bf16* __restrict__ W2t) {
    const int n1 = NEXP * HDIM * FDIM;
    const int n2 = NEXP * ADIM * HDIM;
    int idx = blockIdx.x * blockDim.x + threadIdx.x;
    if (idx < n1) {
        int e = idx / (HDIM * FDIM);
        int r = idx % (HDIM * FDIM);
        int h = r / FDIM;
        int f = r % FDIM;
        W1t[idx] = (__bf16)W1[((size_t)e * FDIM + f) * HDIM + h];
    } else if (idx < n1 + n2) {
        int i = idx - n1;
        int e = i / (ADIM * HDIM);
        int r = i % (ADIM * HDIM);
        int a = r / HDIM;
        int h = r % HDIM;
        W2t[i] = (__bf16)W2[((size_t)e * HDIM + h) * ADIM + a];
    }
}

// Build a 16x32 bf16 A-fragment from a row-major bf16 array (LDS resident).
// Lane layout: lanes 0-15 -> K {k0..k0+15},
// lanes 16-31 -> K {k0+16..k0+31}; row = lane%16.
__device__ __forceinline__ v16bf load_a_frag(const __bf16* p) {
    v8bf lo = *(const v8bf*)p;
    v8bf hi = *(const v8bf*)(p + 8);
    return __builtin_shufflevector(lo, hi, 0,1,2,3,4,5,6,7,8,9,10,11,12,13,14,15);
}

// ---------------- fused MoE policy-net kernel ------------------------------
__launch_bounds__(256, 1)
__global__ void moe_policy_kernel(const float* __restrict__ feat,
                                  const float* __restrict__ Wg,
                                  const float* __restrict__ bg,
                                  const float* __restrict__ b1,
                                  const float* __restrict__ b2,
                                  const __bf16* __restrict__ W1t,
                                  const __bf16* __restrict__ W2t,
                                  float* __restrict__ out) {
    __shared__ __align__(16) __bf16 Xs[M_TILE * XS_STRIDE];   // 133,120 B
    __shared__ __align__(16) __bf16 Hs[M_TILE * HS_STRIDE];   //  67,584 B
    __shared__ float Gp[2][M_TILE][NEXP];                     //   4,096 B
    __shared__ float Gs[M_TILE * NEXP];                       //   2,048 B

    const int tid  = threadIdx.x;
    const int lane = tid & 31;
    const int w    = tid >> 5;        // wave id 0..7
    const int mg   = w & 1;           // GEMM1 row group (64 rows each)
    const int ng   = w >> 1;          // GEMM1 col group (64 cols each)
    const int lr   = lane & 15;       // lane row/col within 16
    const int lh   = lane >> 4;       // lane half
    const long rowBase = (long)blockIdx.x * M_TILE;

    // ---- stage features tile -> bf16 LDS (each element read once from HBM)
    {
        const float4* f4 = (const float4*)(feat + rowBase * FDIM);
        for (int i = tid; i < M_TILE * (FDIM / 4); i += 256) {
            int r = i / (FDIM / 4);
            int c = i % (FDIM / 4);
            float4 v = f4[(size_t)r * (FDIM / 4) + c];
            __bf16* d = &Xs[r * XS_STRIDE + c * 4];
            d[0] = (__bf16)v.x; d[1] = (__bf16)v.y;
            d[2] = (__bf16)v.z; d[3] = (__bf16)v.w;
        }
    }
    __syncthreads();

    // ---- gating, stage 1: all 256 threads compute a 256-deep partial dot.
    // thread t -> row t&127, K half t>>7 (K offset uniform per wave -> s_loads)
    {
        const int grow  = tid & (M_TILE - 1);
        const int ghalf = tid >> 7;
        const int kbeg  = ghalf * (FDIM / 2);
        float l0 = 0.f, l1 = 0.f, l2 = 0.f, l3 = 0.f;
        const __bf16* xr  = &Xs[grow * XS_STRIDE + kbeg];
        const float4* wg4 = (const float4*)Wg;   // Wg[k][0..3]
        for (int k = 0; k < FDIM / 2; k += 8) {
            v8bf xv = *(const v8bf*)(xr + k);
#pragma unroll
            for (int j = 0; j < 8; ++j) {
                float  x = (float)xv[j];
                float4 g = wg4[kbeg + k + j];
                l0 = fmaf(x, g.x, l0); l1 = fmaf(x, g.y, l1);
                l2 = fmaf(x, g.z, l2); l3 = fmaf(x, g.w, l3);
            }
        }
        Gp[ghalf][grow][0] = l0; Gp[ghalf][grow][1] = l1;
        Gp[ghalf][grow][2] = l2; Gp[ghalf][grow][3] = l3;
    }
    __syncthreads();

    // ---- gating, stage 2: softmax over 4 experts (threads 0..127)
    if (tid < M_TILE) {
        float l0 = Gp[0][tid][0] + Gp[1][tid][0] + bg[0];
        float l1 = Gp[0][tid][1] + Gp[1][tid][1] + bg[1];
        float l2 = Gp[0][tid][2] + Gp[1][tid][2] + bg[2];
        float l3 = Gp[0][tid][3] + Gp[1][tid][3] + bg[3];
        float m  = fmaxf(fmaxf(l0, l1), fmaxf(l2, l3));
        float e0 = __expf(l0 - m), e1 = __expf(l1 - m);
        float e2 = __expf(l2 - m), e3 = __expf(l3 - m);
        float inv = 1.0f / (e0 + e1 + e2 + e3);
        Gs[tid * NEXP + 0] = e0 * inv;
        Gs[tid * NEXP + 1] = e1 * inv;
        Gs[tid * NEXP + 2] = e2 * inv;
        Gs[tid * NEXP + 3] = e3 * inv;
    }
    __syncthreads();

    // ---- persistent gate-weighted output accumulators (rows 16*w.., cols 0..31)
    v8f facc0, facc1;
#pragma unroll
    for (int r = 0; r < 8; ++r) { facc0[r] = 0.0f; facc1[r] = 0.0f; }

    const int rowOff = 64 * mg;   // GEMM1 row offset within tile
    const int colOff = 64 * ng;   // GEMM1 col offset within HDIM

    for (int e = 0; e < NEXP; ++e) {
        // ================= GEMM1: h = relu(X @ W1_e + b1_e) =================
        v8f acc[4][4];
#pragma unroll
        for (int mt = 0; mt < 4; ++mt)
#pragma unroll
            for (int nt = 0; nt < 4; ++nt)
#pragma unroll
                for (int r = 0; r < 8; ++r) acc[mt][nt][r] = 0.0f;

        const __bf16* W1e = W1t + (size_t)e * HDIM * FDIM;
        for (int k0 = 0; k0 < FDIM; k0 += 32) {
            v16bf a[4];
#pragma unroll
            for (int mt = 0; mt < 4; ++mt)
                a[mt] = load_a_frag(&Xs[(rowOff + 16 * mt + lr) * XS_STRIDE + k0 + 16 * lh]);
#pragma unroll
            for (int nt = 0; nt < 4; ++nt) {
                // B 32x16: lane holds col (colOff+16*nt+lr), K = k0+16*lh .. +15 contiguous
                const __bf16* bp = W1e + (size_t)(colOff + 16 * nt + lr) * FDIM + k0 + 16 * lh;
                v16bf b = *(const v16bf*)bp;
#pragma unroll
                for (int mt = 0; mt < 4; ++mt)
                    acc[mt][nt] = __builtin_amdgcn_wmma_f32_16x16x32_bf16(
                        false, a[mt], false, b, (short)0, acc[mt][nt], false, false);
            }
        }

        // bias + relu -> bf16 h tile in LDS (C/D layout: row = r + 8*lh, col = lr)
#pragma unroll
        for (int nt = 0; nt < 4; ++nt) {
            float bv = b1[e * HDIM + colOff + 16 * nt + lr];
#pragma unroll
            for (int mt = 0; mt < 4; ++mt) {
#pragma unroll
                for (int r = 0; r < 8; ++r) {
                    float v = fmaxf(acc[mt][nt][r] + bv, 0.0f);
                    Hs[(rowOff + 16 * mt + r + 8 * lh) * HS_STRIDE + colOff + 16 * nt + lr] = (__bf16)v;
                }
            }
        }
        __syncthreads();   // h fully materialized for this expert

        // ================= GEMM2: out_e = h @ W2_e; wave w -> rows 16*w.. ====
        v8f acc2[2];
#pragma unroll
        for (int nt2 = 0; nt2 < 2; ++nt2)
#pragma unroll
            for (int r = 0; r < 8; ++r) acc2[nt2][r] = 0.0f;

        const __bf16* W2e = W2t + (size_t)e * ADIM * HDIM;
        for (int k0 = 0; k0 < HDIM; k0 += 32) {
            v16bf a = load_a_frag(&Hs[(16 * w + lr) * HS_STRIDE + k0 + 16 * lh]);
#pragma unroll
            for (int nt2 = 0; nt2 < 2; ++nt2) {
                const __bf16* bp = W2e + (size_t)(16 * nt2 + lr) * HDIM + k0 + 16 * lh;
                v16bf b = *(const v16bf*)bp;
                acc2[nt2] = __builtin_amdgcn_wmma_f32_16x16x32_bf16(
                    false, a, false, b, (short)0, acc2[nt2], false, false);
            }
        }

        // gate-weighted accumulate: acc += gate[row][e] * (d + b2)
        float bias0 = b2[e * ADIM + lr];
        float bias1 = b2[e * ADIM + 16 + lr];
#pragma unroll
        for (int r = 0; r < 8; ++r) {
            float g = Gs[(16 * w + r + 8 * lh) * NEXP + e];
            facc0[r] = fmaf(g, acc2[0][r] + bias0, facc0[r]);
            facc1[r] = fmaf(g, acc2[1][r] + bias1, facc1[r]);
        }
        __syncthreads();   // safe to overwrite Hs for next expert
    }

    // ---- store action means [128 x 32] (half-wave writes 64B contiguous)
#pragma unroll
    for (int r = 0; r < 8; ++r) {
        long row = rowBase + 16 * w + r + 8 * lh;
        out[row * ADIM + lr]      = facc0[r];
        out[row * ADIM + 16 + lr] = facc1[r];
    }
}

// ---------------------------------------------------------------------------
extern "C" void kernel_launch(void* const* d_in, const int* in_sizes, int n_in,
                              void* d_out, int out_size, void* d_ws, size_t ws_size,
                              hipStream_t stream) {
    const float* feat = (const float*)d_in[0];
    const float* Wg   = (const float*)d_in[1];
    const float* bg   = (const float*)d_in[2];
    const float* W1   = (const float*)d_in[3];
    const float* b1   = (const float*)d_in[4];
    const float* W2   = (const float*)d_in[5];
    const float* b2   = (const float*)d_in[6];
    float* out = (float*)d_out;

    __bf16* W1t = (__bf16*)d_ws;                       // 4*256*512  bf16 = 1 MB
    __bf16* W2t = W1t + (size_t)NEXP * HDIM * FDIM;    // 4*32*256   bf16 = 64 KB

    const int nprep = NEXP * HDIM * FDIM + NEXP * ADIM * HDIM;
    moe_prep_weights<<<(nprep + 255) / 256, 256, 0, stream>>>(W1, W2, W1t, W2t);

    moe_policy_kernel<<<BATCH / M_TILE, 256, 0, stream>>>(
        feat, Wg, bg, b1, b2, W1t, W2t, out);
}
